// CrossGraphDA_15444702396481
// MI455X (gfx1250) — matrix operand order repaired and checked by reading, not compiled, not run on hardware
//
#include <hip/hip_runtime.h>

#define NN   8192
#define KTOP 6

typedef __attribute__((ext_vector_type(16))) _Float16 v16h;
typedef __attribute__((ext_vector_type(8)))  _Float16 v8h;
typedef __attribute__((ext_vector_type(8)))  float    v8f;
typedef __attribute__((ext_vector_type(4)))  float    v4f;

// ---------------- Linear(32->160): h = x @ W1^T + b1 (f32 + f16 copies) ------
__global__ void lin1_kernel(const float* __restrict__ x, const float* __restrict__ W,
                            const float* __restrict__ b, float* __restrict__ h,
                            _Float16* __restrict__ h16) {
  int t = blockIdx.x * blockDim.x + threadIdx.x;
  if (t >= NN * 160) return;
  int c = t % 160, i = t / 160;
  const float* xr = x + i * 32;
  const float* wr = W + c * 32;
  float acc = b[c];
#pragma unroll
  for (int k = 0; k < 32; ++k) acc = fmaf(xr[k], wr[k], acc);
  h[t] = acc;
  h16[t] = (_Float16)acc;
}

// ---------------- f32 -> f16 convert (weights) -------------------------------
__global__ void cvt16_kernel(const float* __restrict__ W, _Float16* __restrict__ W16, int n) {
  int t = blockIdx.x * blockDim.x + threadIdx.x;
  if (t < n) W16[t] = (_Float16)W[t];
}

// ---------------- Q/K projections via WMMA -----------------------------------
// C[16x16] tiles of Q = h16 @ Wq^T and K = h16 @ Wk^T. Row-major [out][in] f16
// weights give exactly the K^T B-tile addressing: lane = output column,
// 16 contiguous K halfs at kc*32 + half*16. One shared A fragment per block
// feeds both Q and K WMMAs. ct-loop is wave-uniform => EXEC all ones.
__global__ __launch_bounds__(128) void proj_wmma_kernel(
    const _Float16* __restrict__ H,
    const _Float16* __restrict__ Wq16, const float* __restrict__ bq,
    const _Float16* __restrict__ Wk16, const float* __restrict__ bk,
    _Float16* __restrict__ Qout, _Float16* __restrict__ Kout) {
  const int lane = threadIdx.x & 31;
  const int wave = threadIdx.x >> 5;
  const int half = lane >> 4;
  const int m16  = lane & 15;
  const int rowbase = blockIdx.x * 16;

  v16h a[5];
#pragma unroll
  for (int kc = 0; kc < 5; ++kc) {
    const _Float16* p = H + (size_t)(rowbase + m16) * 160 + kc * 32;
    v8h lo = *(const v8h*)(p + half * 16);
    v8h hi = *(const v8h*)(p + half * 16 + 8);
#pragma unroll
    for (int e = 0; e < 8; ++e) { a[kc][e] = lo[e]; a[kc][8 + e] = hi[e]; }
  }

  for (int ct = wave; ct < 10; ct += 4) {   // 160/16 = 10 column tiles
    const int cb = ct * 16;
    v8f cq = {}, ck = {};
#pragma unroll
    for (int kc = 0; kc < 5; ++kc) {
      v16h bqv = *(const v16h*)(Wq16 + (size_t)(cb + m16) * 160 + kc * 32 + half * 16);
      v16h bkv = *(const v16h*)(Wk16 + (size_t)(cb + m16) * 160 + kc * 32 + half * 16);
      cq = __builtin_amdgcn_wmma_f32_16x16x32_f16(false, a[kc], false, bqv, (short)0, cq, false, false);
      ck = __builtin_amdgcn_wmma_f32_16x16x32_f16(false, a[kc], false, bkv, (short)0, ck, false, false);
    }
    float biq = bq[cb + m16], bik = bk[cb + m16];
#pragma unroll
    for (int r = 0; r < 8; ++r) {
      int row = rowbase + half * 8 + r;
      Qout[(size_t)row * 160 + cb + m16] = (_Float16)(cq[r] + biq);
      Kout[(size_t)row * 160 + cb + m16] = (_Float16)(ck[r] + bik);
    }
  }
}

// ---------------- WMMA scores (Q @ K^T) + running top-6 per row --------------
// 128 threads = 4 waves per block, 32 query rows per block (two A fragments per
// wave share every B fragment -> 10 WMMA per loaded K^T tile, halving L2
// traffic vs 16 rows). 16x64-per-fragment f32 tiles double-buffered through LDS
// (one barrier/iter); 32 rows x 4 col-groups update per-thread top-6 with a
// max16 pre-filter. Softmax + 1/sqrt(160) skipped: monotonic, top-k invariant.
__global__ __launch_bounds__(128) void scores_topk_kernel(
    const _Float16* __restrict__ Q, const _Float16* __restrict__ Kt,
    int* __restrict__ topk) {
  __shared__ float sTile[2][32][68];
  __shared__ float candV[32][24];
  __shared__ int   candI[32][24];

  const int lane = threadIdx.x & 31;
  const int wave = threadIdx.x >> 5;
  const int half = lane >> 4;
  const int m16  = lane & 15;
  const int rowbase = blockIdx.x * 32;

  v16h a0[5], a1[5];
#pragma unroll
  for (int kc = 0; kc < 5; ++kc) {
    const _Float16* p0 = Q + (size_t)(rowbase + m16) * 160 + kc * 32;
    const _Float16* p1 = p0 + (size_t)16 * 160;
    v8h lo0 = *(const v8h*)(p0 + half * 16);
    v8h hi0 = *(const v8h*)(p0 + half * 16 + 8);
    v8h lo1 = *(const v8h*)(p1 + half * 16);
    v8h hi1 = *(const v8h*)(p1 + half * 16 + 8);
#pragma unroll
    for (int e = 0; e < 8; ++e) {
      a0[kc][e] = lo0[e]; a0[kc][8 + e] = hi0[e];
      a1[kc][e] = lo1[e]; a1[kc][8 + e] = hi1[e];
    }
  }

  const int srow = threadIdx.x >> 2;  // 0..31
  const int sgrp = threadIdx.x & 3;   // 0..3, 16 columns each
  float bv[6];
  int   bi[6];
#pragma unroll
  for (int k = 0; k < 6; ++k) { bv[k] = -3.0e38f; bi[k] = 0; }

  for (int it = 0; it < NN / 64; ++it) {
    const int jb = it * 64 + wave * 16;
    v8f c0 = {}, c1 = {};
#pragma unroll
    for (int kc = 0; kc < 5; ++kc) {
      v16h bt = *(const v16h*)(Kt + (size_t)(jb + m16) * 160 + kc * 32 + half * 16);
      c0 = __builtin_amdgcn_wmma_f32_16x16x32_f16(false, a0[kc], false, bt,
                                                  (short)0, c0, false, false);
      c1 = __builtin_amdgcn_wmma_f32_16x16x32_f16(false, a1[kc], false, bt,
                                                  (short)0, c1, false, false);
    }
    if (it + 1 < NN / 64)  // prefetch next column tile's K^T rows (L2 -> WGP$)
      __builtin_prefetch(Kt + (size_t)(jb + 64 + m16) * 160, 0, 1);

#pragma unroll
    for (int r = 0; r < 8; ++r) {
      sTile[it & 1][half * 8 + r][wave * 16 + m16]      = c0[r];
      sTile[it & 1][16 + half * 8 + r][wave * 16 + m16] = c1[r];
    }
    __syncthreads();  // double-buffered: one barrier per iteration is safe

    const float* rowp = &sTile[it & 1][srow][sgrp * 16];
    v4f y0 = *(const v4f*)rowp;
    v4f y1 = *(const v4f*)(rowp + 4);
    v4f y2 = *(const v4f*)(rowp + 8);
    v4f y3 = *(const v4f*)(rowp + 12);
    float m0 = fmaxf(fmaxf(fmaxf(y0[0], y0[1]), fmaxf(y0[2], y0[3])),
                     fmaxf(fmaxf(y1[0], y1[1]), fmaxf(y1[2], y1[3])));
    float m1 = fmaxf(fmaxf(fmaxf(y2[0], y2[1]), fmaxf(y2[2], y2[3])),
                     fmaxf(fmaxf(y3[0], y3[1]), fmaxf(y3[2], y3[3])));
    if (fmaxf(m0, m1) > bv[5]) {  // rare after warm-up
      const int j0 = it * 64 + sgrp * 16;
#pragma unroll
      for (int u = 0; u < 16; ++u) {
        float v = (u < 4) ? y0[u] : (u < 8) ? y1[u - 4] : (u < 12) ? y2[u - 8] : y3[u - 12];
        if (v > bv[5]) {
          int j = j0 + u;
#pragma unroll
          for (int q = 0; q < 6; ++q) {   // bubble-insert, const indices
            bool gt = (v > bv[q]);
            float tv = bv[q]; int ti = bi[q];
            if (gt) { bv[q] = v; bi[q] = j; v = tv; j = ti; }
          }
        }
      }
    }
  }

  __syncthreads();
#pragma unroll
  for (int k = 0; k < 6; ++k) {
    candV[srow][sgrp * 6 + k] = bv[k];
    candI[srow][sgrp * 6 + k] = bi[k];
  }
  __syncthreads();

  if (threadIdx.x < 32) {
    int row = threadIdx.x;
    unsigned used = 0u;
    for (int k = 0; k < 6; ++k) {
      float best = -3.4e38f; int besti = 0x7fffffff; int bslot = 0;
      for (int s = 0; s < 24; ++s) {
        if (used & (1u << s)) continue;
        float v = candV[row][s]; int ji = candI[row][s];
        if (v > best || (v == best && ji < besti)) { best = v; besti = ji; bslot = s; }
      }
      used |= (1u << bslot);
      topk[(rowbase + row) * KTOP + k] = besti;
    }
  }
}

// ---------------- SAGE neighbor aggregation (masked mean via atomics) --------
__global__ void agg_kernel(const float* __restrict__ X, const int* __restrict__ idx,
                           float* __restrict__ agg, float* __restrict__ cnt, int Cin) {
  long long t = (long long)blockIdx.x * blockDim.x + threadIdx.x;
  long long total = (long long)NN * KTOP * Cin;
  if (t >= total) return;
  int c = (int)(t % Cin);
  int e = (int)(t / Cin);
  int j = e / KTOP;
  int dst = idx[e];
  if (j == dst) return;   // w = (src != dst)
  atomicAdd(&agg[(long long)dst * Cin + c], X[(long long)j * Cin + c]);
  if (c == 0) atomicAdd(&cnt[dst], 1.0f);
}

// ---------------- SAGE linear: out = (agg/max(cnt,1))@Wl^T + bl + X@Wr^T -----
__global__ void sage_lin_kernel(const float* __restrict__ agg, const float* __restrict__ cnt,
                                const float* __restrict__ X,
                                const float* __restrict__ Wl, const float* __restrict__ bl,
                                const float* __restrict__ Wr,
                                float* __restrict__ out, int Cin, int Cout) {
  int t = blockIdx.x * blockDim.x + threadIdx.x;
  if (t >= NN * Cout) return;
  int o = t % Cout, i = t / Cout;
  float inv = 1.0f / fmaxf(cnt[i], 1.0f);
  const float* ar  = agg + (long long)i * Cin;
  const float* xr  = X   + (long long)i * Cin;
  const float* wlr = Wl  + (long long)o * Cin;
  const float* wrr = Wr  + (long long)o * Cin;
  float acc = bl[o];
  for (int c = 0; c < Cin; ++c)
    acc = fmaf(ar[c] * inv, wlr[c], fmaf(xr[c], wrr[c], acc));
  out[t] = acc;
}

// ---------------- BatchNorm stats (sum, sumsq per column) --------------------
__global__ void bn_sums_kernel(const float* __restrict__ X, float* __restrict__ sums, int C) {
  __shared__ float ls[128];
  for (int k = threadIdx.x; k < 2 * C; k += blockDim.x) ls[k] = 0.f;
  __syncthreads();
  for (long long t = (long long)blockIdx.x * blockDim.x + threadIdx.x;
       t < (long long)NN * C; t += (long long)gridDim.x * blockDim.x) {
    float v = X[t];
    int c = (int)(t % C);
    atomicAdd(&ls[c], v);
    atomicAdd(&ls[C + c], v * v);
  }
  __syncthreads();
  for (int k = threadIdx.x; k < 2 * C; k += blockDim.x) atomicAdd(&sums[k], ls[k]);
}

// ---------------- BN (training-mode, biased var) + ReLU ----------------------
__global__ void bn_apply_kernel(const float* __restrict__ X, const float* __restrict__ sums,
                                const float* __restrict__ g, const float* __restrict__ b,
                                float* __restrict__ out, int C) {
  int t = blockIdx.x * blockDim.x + threadIdx.x;
  if (t >= NN * C) return;
  int c = t % C;
  float mu  = sums[c] * (1.0f / NN);
  float var = sums[C + c] * (1.0f / NN) - mu * mu;
  float y = (X[t] - mu) * rsqrtf(var + 1e-5f) * g[c] + b[c];
  out[t] = fmaxf(y, 0.f);
}

// ---------------- G = [f1|f2]@convW^T + b; column sums of 2*x3-G, 2*x4-G -----
__global__ void final_g_kernel(const float* __restrict__ f1, const float* __restrict__ f2,
                               const float* __restrict__ x3, const float* __restrict__ x4,
                               const float* __restrict__ convW, const float* __restrict__ convb,
                               float* __restrict__ s34) {
  int t = blockIdx.x * blockDim.x + threadIdx.x;
  if (t >= NN * 32) return;
  int c = t % 32, i = t / 32;
  const float* w = convW + c * 64;
  float acc = convb[c];
#pragma unroll
  for (int k = 0; k < 32; ++k)
    acc = fmaf(f1[i * 32 + k], w[k], fmaf(f2[i * 32 + k], w[32 + k], acc));
  atomicAdd(&s34[c],      2.0f * x3[t] - acc);
  atomicAdd(&s34[32 + c], 2.0f * x4[t] - acc);
}

__global__ void final_out_kernel(const float* __restrict__ s34, float* __restrict__ out) {
  if (threadIdx.x == 0 && blockIdx.x == 0) {
    float acc = 0.f;
    for (int c = 0; c < 32; ++c) {
      float d = (s34[c] - s34[32 + c]) * (1.0f / NN);
      acc = fmaf(d, d, acc);
    }
    out[0] = acc;
  }
}

// =============================================================================
extern "C" void kernel_launch(void* const* d_in, const int* in_sizes, int n_in,
                              void* d_out, int out_size, void* d_ws, size_t ws_size,
                              hipStream_t stream) {
  (void)in_sizes; (void)n_in; (void)out_size; (void)ws_size;
  const float* x1    = (const float*)d_in[0];
  const float* x2    = (const float*)d_in[1];
  const float* x3    = (const float*)d_in[2];
  const float* x4    = (const float*)d_in[3];
  const float* w1_W  = (const float*)d_in[4];
  const float* w1_b  = (const float*)d_in[5];
  const float* wq_W  = (const float*)d_in[6];
  const float* wq_b  = (const float*)d_in[7];
  const float* wk_W  = (const float*)d_in[8];
  const float* wk_b  = (const float*)d_in[9];
  const float* s1Wl  = (const float*)d_in[10];
  const float* s1bl  = (const float*)d_in[11];
  const float* s1Wr  = (const float*)d_in[12];
  const float* bn1g  = (const float*)d_in[13];
  const float* bn1b  = (const float*)d_in[14];
  const float* s2Wl  = (const float*)d_in[15];
  const float* s2bl  = (const float*)d_in[16];
  const float* s2Wr  = (const float*)d_in[17];
  const float* bn2g  = (const float*)d_in[18];
  const float* bn2b  = (const float*)d_in[19];
  const float* convW = (const float*)d_in[20];
  const float* convb = (const float*)d_in[21];

  char* wsb = (char*)d_ws;
  size_t off = 0;
  auto take = [&](size_t bytes) -> char* {
    char* p = wsb + off;
    off = (off + bytes + 255) & ~(size_t)255;
    return p;
  };
  float*    h    = (float*)take((size_t)NN * 160 * 4);
  _Float16* h16  = (_Float16*)take((size_t)NN * 160 * 2);
  _Float16* q16  = (_Float16*)take((size_t)NN * 160 * 2);
  _Float16* k16  = (_Float16*)take((size_t)NN * 160 * 2);
  _Float16* wq16 = (_Float16*)take((size_t)160 * 160 * 2);
  _Float16* wk16 = (_Float16*)take((size_t)160 * 160 * 2);
  int*      idx  = (int*)take((size_t)NN * KTOP * 4);
  float*    agg  = (float*)take((size_t)NN * 160 * 4);
  float*    cnt  = (float*)take((size_t)NN * 4);
  float*    sout = (float*)take((size_t)NN * 64 * 4);
  float*    g1   = (float*)take((size_t)NN * 64 * 4);
  float*    f0   = (float*)take((size_t)NN * 32 * 4);
  float*    f1b  = (float*)take((size_t)NN * 32 * 4);
  float*    stats = (float*)take(2 * 64 * 4);
  float*    s34   = (float*)take(64 * 4);
  float*    fout[2] = { f0, f1b };

  const int TB = 256;

  // One-time weight conversions (graph-capture safe, deterministic).
  cvt16_kernel<<<(160 * 160 + TB - 1) / TB, TB, 0, stream>>>(wq_W, wq16, 160 * 160);
  cvt16_kernel<<<(160 * 160 + TB - 1) / TB, TB, 0, stream>>>(wk_W, wk16, 160 * 160);

  for (int b = 0; b < 2; ++b) {
    const float* x = b ? x2 : x1;
    lin1_kernel<<<(NN * 160 + TB - 1) / TB, TB, 0, stream>>>(x, w1_W, w1_b, h, h16);
    proj_wmma_kernel<<<NN / 16, 128, 0, stream>>>(h16, wq16, wq_b, wk16, wk_b, q16, k16);
    scores_topk_kernel<<<NN / 32, 128, 0, stream>>>(q16, k16, idx);

    // SAGE1: 160 -> 64
    hipMemsetAsync(agg, 0, (size_t)NN * 160 * 4, stream);
    hipMemsetAsync(cnt, 0, (size_t)NN * 4, stream);
    {
      long long tot = (long long)NN * KTOP * 160;
      agg_kernel<<<(unsigned)((tot + TB - 1) / TB), TB, 0, stream>>>(h, idx, agg, cnt, 160);
    }
    sage_lin_kernel<<<(NN * 64 + TB - 1) / TB, TB, 0, stream>>>(agg, cnt, h, s1Wl, s1bl, s1Wr, sout, 160, 64);
    hipMemsetAsync(stats, 0, 2 * 64 * 4, stream);
    bn_sums_kernel<<<512, TB, 0, stream>>>(sout, stats, 64);
    bn_apply_kernel<<<(NN * 64 + TB - 1) / TB, TB, 0, stream>>>(sout, stats, bn1g, bn1b, g1, 64);

    // SAGE2: 64 -> 32
    hipMemsetAsync(agg, 0, (size_t)NN * 64 * 4, stream);
    hipMemsetAsync(cnt, 0, (size_t)NN * 4, stream);
    {
      long long tot = (long long)NN * KTOP * 64;
      agg_kernel<<<(unsigned)((tot + TB - 1) / TB), TB, 0, stream>>>(g1, idx, agg, cnt, 64);
    }
    sage_lin_kernel<<<(NN * 32 + TB - 1) / TB, TB, 0, stream>>>(agg, cnt, g1, s2Wl, s2bl, s2Wr, sout, 64, 32);
    hipMemsetAsync(stats, 0, 2 * 64 * 4, stream);
    bn_sums_kernel<<<512, TB, 0, stream>>>(sout, stats, 32);
    bn_apply_kernel<<<(NN * 32 + TB - 1) / TB, TB, 0, stream>>>(sout, stats, bn2g, bn2b, fout[b], 32);
  }

  hipMemsetAsync(s34, 0, 64 * 4, stream);
  final_g_kernel<<<(NN * 32 + TB - 1) / TB, TB, 0, stream>>>(fout[0], fout[1], x3, x4, convW, convb, s34);
  final_out_kernel<<<1, 32, 0, stream>>>(s34, (float*)d_out);
}
